// CommAgent_60172491817018
// MI455X (gfx1250) — matrix operand order repaired and checked by reading, not compiled, not run on hardware
//
#include <hip/hip_runtime.h>
#include <math.h>

// ---------------------------------------------------------------------------
// Fully-fused CommAgent forward for MI455X (gfx1250, wave32, WMMA).
//
//   x     = relu(X @ W1^T + b1)                       (32x512)·(512x256)
//   h_rnn = GRU(x, h_in; rnn_W*)                      two (32x256)·(256x768)
//   4x comm:  c = (nb @ h)/n ; h = GRU(h, c; comm_W*) (32x32)·(32x256) + gates
//   q     = h @ W2^T + b2                             (32x256)·(256x16)
//
// One workgroup (256 threads = 8 wave32) owns one 32-agent group (32 rows).
// All activations live in LDS for the whole pipeline; weights are converted
// once to bf16 in d_ws (L2-resident, 1.85 MB) and consumed directly as WMMA
// B-fragments (rows of W are [n][k] = exactly the B layout).
// Matrix math: v_wmma_f32_16x16x32_bf16, f32 accumulate.
// Elementwise GRU uses v_exp_f32/v_rcp_f32 forms that co-execute with XDL.
// ---------------------------------------------------------------------------

#define N_AGENTS 32
#define HDIM     256
#define INPF     512
#define NACT     16
#define NROWS    65536
#define HLD      (HDIM + 8)      // padded LDS row (bf16 elems) -> conflict-free b128
#define XLD      (INPF + 8)
#define TLD      (N_AGENTS + 8)

typedef __bf16 bf16;
typedef __attribute__((ext_vector_type(16))) __bf16 v16bf;
typedef __attribute__((ext_vector_type(8)))  float  v8f;

struct Frag32B { uint4 a, b; };

__device__ __forceinline__ unsigned int f2bf_bits(float f) {
  unsigned int u = __builtin_bit_cast(unsigned int, f);
  u += 0x7FFFu + ((u >> 16) & 1u);            // round-to-nearest-even
  return u >> 16;
}
__device__ __forceinline__ bf16 f2bf(float f) {
  unsigned short b = (unsigned short)f2bf_bits(f);
  return __builtin_bit_cast(bf16, b);
}
// sigmoid = v_rcp(1 + v_exp(-x)); tanh = 2*sigmoid(2x) - 1.
// TRANS32 ops co-execute with XDL WMMA on CDNA5.
__device__ __forceinline__ float sigmoid_fast(float x) {
  return __builtin_amdgcn_rcpf(1.0f + __expf(-x));
}
__device__ __forceinline__ float tanh_fast(float x) {
  return fmaf(2.0f, __builtin_amdgcn_rcpf(1.0f + __expf(-2.0f * x)), -1.0f);
}

// A fragment (16x32 bf16, M x K), source row-major [m][k], ld in elements.
// ISA 7.12.2: lanes 0-15 m=0..15; V0..3 K=8*hi.., V4..7 K=16+8*hi..
__device__ __forceinline__ v16bf loadFragA(const bf16* base, int ld, int m0,
                                           int k0, int lane) {
  int m  = m0 + (lane & 15);
  int hi = lane >> 4;
  const bf16* p = base + (size_t)m * ld + k0 + 8 * hi;
  Frag32B s;
  s.a = *(const uint4*)(p);        // K = k0+8hi .. +7
  s.b = *(const uint4*)(p + 16);   // K = k0+16+8hi .. +7
  return __builtin_bit_cast(v16bf, s);
}

// B fragment (32x16 bf16, K x N), source [n][k] row-major (k contiguous).
// ISA 7.12.2: V0..3 K=8*hi.., V4..7 K=16+8*hi.. (matches A fragment).
__device__ __forceinline__ v16bf loadFragB(const bf16* base, int ld, int n0,
                                           int k0, int lane) {
  int n  = n0 + (lane & 15);
  int hi = lane >> 4;
  const bf16* p = base + (size_t)n * ld + k0 + 8 * hi;
  Frag32B s;
  s.a = *(const uint4*)(p);        // K = 8hi .. +7
  s.b = *(const uint4*)(p + 16);   // K = 16+8hi .. +7
  return __builtin_bit_cast(v16bf, s);
}

__device__ __forceinline__ v8f wmma_bf16(v16bf a, v16bf b, v8f c) {
  return __builtin_amdgcn_wmma_f32_16x16x32_bf16(false, a, false, b,
                                                 (short)0, c, false, false);
}

// One GRU: gi = Agi @ Wih^T, gh = Agh @ Whh^T, elementwise fused in C-layout.
// Wave owns N-tiles {2w, 2w+1}; both M-tiles (32 rows). 24 f32 accum tiles.
__device__ __forceinline__ void gru_stage(
    const bf16* Agi, int ldgi,           // A for input gates
    const bf16* Agh, int ldgh,           // A for hidden gates
    const bf16* __restrict__ Wih, const bf16* __restrict__ Whh,  // [768][256] bf16
    const float* __restrict__ bih, const float* __restrict__ bhh,
    const float* __restrict__ hargG,     // f32 h_arg (rnn path), pre-offset to r0; or null
    const v8f* cargs,                    // h_arg tiles in regs (comm path); or null
    bf16* hOut, int ldho,                // new h, row-major bf16
    bf16* hTout, int ldht,               // new h, transposed [feat][agent]
    float* __restrict__ hOutG,           // f32 global h_rnn out (pre-offset) or null
    int wave, int lane) {
  int hi = lane >> 4;
#pragma unroll
  for (int nti = 0; nti < 2; ++nti) {
    int nt   = wave * 2 + nti;
    int ncol = nt * 16 + (lane & 15);
    v8f gi[3][2], gh[3][2];              // [gate][mtile]
#pragma unroll
    for (int g = 0; g < 3; ++g)
#pragma unroll
      for (int mt = 0; mt < 2; ++mt) { gi[g][mt] = (v8f){}; gh[g][mt] = (v8f){}; }

#pragma unroll 2
    for (int kt = 0; kt < 8; ++kt) {
      int k0 = kt * 32;
      v16bf Bi[3], Bh[3];
#pragma unroll
      for (int g = 0; g < 3; ++g) {
        Bi[g] = loadFragB(Wih, HDIM, g * HDIM + nt * 16, k0, lane);
        Bh[g] = loadFragB(Whh, HDIM, g * HDIM + nt * 16, k0, lane);
      }
      v16bf Ai0 = loadFragA(Agi, ldgi, 0,  k0, lane);
      v16bf Ai1 = loadFragA(Agi, ldgi, 16, k0, lane);
      v16bf Ah0 = loadFragA(Agh, ldgh, 0,  k0, lane);
      v16bf Ah1 = loadFragA(Agh, ldgh, 16, k0, lane);
#pragma unroll
      for (int g = 0; g < 3; ++g) {
        gi[g][0] = wmma_bf16(Ai0, Bi[g], gi[g][0]);
        gi[g][1] = wmma_bf16(Ai1, Bi[g], gi[g][1]);
        gh[g][0] = wmma_bf16(Ah0, Bh[g], gh[g][0]);
        gh[g][1] = wmma_bf16(Ah1, Bh[g], gh[g][1]);
      }
    }

    float bir = bih[ncol], biz = bih[HDIM + ncol], bin = bih[2 * HDIM + ncol];
    float bhr = bhh[ncol], bhz = bhh[HDIM + ncol], bhn = bhh[2 * HDIM + ncol];
#pragma unroll
    for (int mt = 0; mt < 2; ++mt) {
      float hnew[8];
#pragma unroll
      for (int v = 0; v < 8; ++v) {
        int m = mt * 16 + v + 8 * hi;    // row in 32-block (C-layout)
        float harg = hargG ? hargG[(size_t)m * HDIM + ncol]
                           : cargs[nti * 2 + mt][v];
        float r = sigmoid_fast(gi[0][mt][v] + bir + gh[0][mt][v] + bhr);
        float z = sigmoid_fast(gi[1][mt][v] + biz + gh[1][mt][v] + bhz);
        float n = tanh_fast(gi[2][mt][v] + bin + r * (gh[2][mt][v] + bhn));
        hnew[v] = (1.0f - z) * n + z * harg;
        hOut[(size_t)m * ldho + ncol] = f2bf(hnew[v]);
        if (hOutG) hOutG[(size_t)m * HDIM + ncol] = hnew[v];
      }
      // Transposed copy: this lane's 8 values are hT[ncol][mt*16+8hi .. +7]
      // -> one packed ds_store_b128 instead of 8 b16 scatters.
      uint4 pk;
      pk.x = f2bf_bits(hnew[0]) | (f2bf_bits(hnew[1]) << 16);
      pk.y = f2bf_bits(hnew[2]) | (f2bf_bits(hnew[3]) << 16);
      pk.z = f2bf_bits(hnew[4]) | (f2bf_bits(hnew[5]) << 16);
      pk.w = f2bf_bits(hnew[6]) | (f2bf_bits(hnew[7]) << 16);
      *(uint4*)&hTout[(size_t)ncol * ldht + mt * 16 + 8 * hi] = pk;
    }
  }
}

__global__ void __launch_bounds__(256)
comm_agent_fused(const float* __restrict__ inputs,
                 const float* __restrict__ hidden,
                 const float* __restrict__ b1,
                 const float* __restrict__ rbih, const float* __restrict__ rbhh,
                 const float* __restrict__ cbih, const float* __restrict__ cbhh,
                 const float* __restrict__ b2,
                 const bf16* __restrict__ W1b,
                 const bf16* __restrict__ rWihb, const bf16* __restrict__ rWhhb,
                 const bf16* __restrict__ cWihb, const bf16* __restrict__ cWhhb,
                 const bf16* __restrict__ W2b,
                 float* __restrict__ out) {
  __shared__ bf16  Xb[N_AGENTS][XLD];       // inputs tile, bf16
  __shared__ bf16  xA[N_AGENTS][HLD];       // relu(X W1^T)
  __shared__ bf16  h0[N_AGENTS][HLD];       // h ping
  __shared__ bf16  h1[N_AGENTS][HLD];       // h pong
  __shared__ bf16  cAsh[N_AGENTS][HLD];     // comm message c
  __shared__ bf16  hT[HDIM][TLD];           // h transposed [feat][agent]
  __shared__ bf16  nbm[N_AGENTS][TLD];      // neighbor matrix 32x32
  __shared__ float recipN[N_AGENTS];        // 1 / n_neighbor

  const int tid  = threadIdx.x;
  const int lane = tid & 31;
  const int wave = tid >> 5;
  const int r0   = blockIdx.x * N_AGENTS;
  const size_t QOFF = (size_t)NROWS * NACT;

  // ---- Stage 0: stage inputs + hidden into LDS (bf16), build nb + 1/n ----
  {
    const float4* src = (const float4*)(inputs + (size_t)r0 * INPF);
#pragma unroll 1
    for (int j = tid; j < (N_AGENTS * INPF) / 4; j += 256) {
      float4 v = src[j];
      int row = j >> 7, cq = (j & 127) << 2;          // 128 float4 per row
      unsigned long long pk = (unsigned long long)f2bf_bits(v.x)
                            | ((unsigned long long)f2bf_bits(v.y) << 16)
                            | ((unsigned long long)f2bf_bits(v.z) << 32)
                            | ((unsigned long long)f2bf_bits(v.w) << 48);
      *(unsigned long long*)&Xb[row][cq] = pk;
    }
    const float4* hs = (const float4*)(hidden + (size_t)r0 * HDIM);
#pragma unroll 1
    for (int j = tid; j < (N_AGENTS * HDIM) / 4; j += 256) {
      float4 v = hs[j];
      int row = j >> 6, cq = (j & 63) << 2;           // 64 float4 per row
      unsigned long long pk = (unsigned long long)f2bf_bits(v.x)
                            | ((unsigned long long)f2bf_bits(v.y) << 16)
                            | ((unsigned long long)f2bf_bits(v.z) << 32)
                            | ((unsigned long long)f2bf_bits(v.w) << 48);
      *(unsigned long long*)&h0[row][cq] = pk;
    }
    // neighbor matrix: obs[:, 260 + 8k], k=0..30; col = k + (k >= i)
    int i = tid >> 3, sub = tid & 7;
    const float* orow = inputs + (size_t)(r0 + i) * INPF + 260;
    if (sub == 0) nbm[i][i] = f2bf(0.0f);
#pragma unroll 1
    for (int k = sub; k < 31; k += 8) {
      float v = orow[k * 8];
      nbm[i][k + (k >= i ? 1 : 0)] = f2bf(v);
    }
    if (tid < N_AGENTS) {
      const float* row = inputs + (size_t)(r0 + tid) * INPF + 260;
      float s = 0.0f;
#pragma unroll 1
      for (int k = 0; k < 31; ++k) s += row[k * 8];
      recipN[tid] = 1.0f / s;
    }
  }
  __syncthreads();

  // ---- Stage 1: x = relu(X @ W1^T + b1) -> xA ----
  {
    int hi = lane >> 4;
#pragma unroll
    for (int nti = 0; nti < 2; ++nti) {
      int nt = wave * 2 + nti;
      int ncol = nt * 16 + (lane & 15);
      v8f a0 = (v8f){}, a1 = (v8f){};
#pragma unroll 2
      for (int kt = 0; kt < 16; ++kt) {
        int k0 = kt * 32;
        v16bf B  = loadFragB(W1b, INPF, nt * 16, k0, lane);
        v16bf A0 = loadFragA(&Xb[0][0], XLD, 0,  k0, lane);
        v16bf A1 = loadFragA(&Xb[0][0], XLD, 16, k0, lane);
        a0 = wmma_bf16(A0, B, a0);
        a1 = wmma_bf16(A1, B, a1);
      }
      float bias = b1[ncol];
#pragma unroll
      for (int v = 0; v < 8; ++v) {
        float x0 = fmaxf(a0[v] + bias, 0.0f);
        float x1 = fmaxf(a1[v] + bias, 0.0f);
        xA[v + 8 * hi][ncol]      = f2bf(x0);
        xA[16 + v + 8 * hi][ncol] = f2bf(x1);
      }
    }
  }
  __syncthreads();

  // ---- Stage 2: rnn GRU: h_rnn = GRU(x, h_in) -> h1, hT, d_out[h section] ----
  gru_stage(&xA[0][0], HLD, &h0[0][0], HLD, rWihb, rWhhb, rbih, rbhh,
            hidden + (size_t)r0 * HDIM, nullptr,
            &h1[0][0], HLD, &hT[0][0], TLD,
            out + QOFF + (size_t)r0 * HDIM, wave, lane);
  __syncthreads();

  // ---- Stage 3: 4 comm steps ----
  bf16* hcur = &h1[0][0];
  bf16* hnxt = &h0[0][0];
#pragma unroll 1
  for (int step = 0; step < 4; ++step) {
    // c = (nb @ h) / n  — single 16x16x32 WMMA per tile, B from transposed hT
    v8f carr[4];
    int hi = lane >> 4;
#pragma unroll
    for (int nti = 0; nti < 2; ++nti) {
      int nt = wave * 2 + nti;
      int ncol = nt * 16 + (lane & 15);
      v16bf Bt = loadFragB(&hT[0][0], TLD, nt * 16, 0, lane);
#pragma unroll
      for (int mt = 0; mt < 2; ++mt) {
        v16bf An = loadFragA(&nbm[0][0], TLD, mt * 16, 0, lane);
        v8f acc = (v8f){};
        acc = wmma_bf16(An, Bt, acc);
#pragma unroll
        for (int v = 0; v < 8; ++v) acc[v] *= recipN[mt * 16 + v + 8 * hi];
        carr[nti * 2 + mt] = acc;
#pragma unroll
        for (int v = 0; v < 8; ++v)
          cAsh[mt * 16 + v + 8 * hi][ncol] = f2bf(acc[v]);
      }
    }
    __syncthreads();
    // h = GRU(h, c): gi from h, gh from c, h_arg = c (kept in registers)
    gru_stage(hcur, HLD, &cAsh[0][0], HLD, cWihb, cWhhb, cbih, cbhh,
              nullptr, carr, hnxt, HLD, &hT[0][0], TLD, nullptr, wave, lane);
    bf16* t = hcur; hcur = hnxt; hnxt = t;
    __syncthreads();
  }

  // ---- Stage 4: q = h @ W2^T + b2 (only 2 tiles; waves 0,1) ----
  if (wave < 2) {
    int mt = wave, hi = lane >> 4, n = lane & 15;
    v8f acc = (v8f){};
#pragma unroll 2
    for (int kt = 0; kt < 8; ++kt) {
      int k0 = kt * 32;
      v16bf A = loadFragA(hcur, HLD, mt * 16, k0, lane);
      v16bf B = loadFragB(W2b, HDIM, 0, k0, lane);
      acc = wmma_bf16(A, B, acc);
    }
    float bias = b2[n];
#pragma unroll
    for (int v = 0; v < 8; ++v) {
      int m = mt * 16 + v + 8 * hi;
      out[(size_t)(r0 + m) * NACT + n] = acc[v] + bias;
    }
  }
}

// Prologue: weights f32 -> bf16 in workspace (stays L2-resident).
__global__ void cvt_f32_to_bf16(const float* __restrict__ s,
                                bf16* __restrict__ d, int n) {
  for (int i = blockIdx.x * blockDim.x + threadIdx.x; i < n;
       i += gridDim.x * blockDim.x)
    d[i] = f2bf(s[i]);
}

extern "C" void kernel_launch(void* const* d_in, const int* in_sizes, int n_in,
                              void* d_out, int out_size, void* d_ws,
                              size_t ws_size, hipStream_t stream) {
  const float* inputs = (const float*)d_in[0];
  const float* hidden = (const float*)d_in[1];
  const float* W1     = (const float*)d_in[2];
  const float* b1     = (const float*)d_in[3];
  const float* rWih   = (const float*)d_in[4];
  const float* rWhh   = (const float*)d_in[5];
  const float* rbih   = (const float*)d_in[6];
  const float* rbhh   = (const float*)d_in[7];
  const float* cWih   = (const float*)d_in[8];
  const float* cWhh   = (const float*)d_in[9];
  const float* cbih   = (const float*)d_in[10];
  const float* cbhh   = (const float*)d_in[11];
  const float* W2     = (const float*)d_in[12];
  const float* b2     = (const float*)d_in[13];
  float* out = (float*)d_out;

  bf16* ws    = (bf16*)d_ws;
  bf16* W1b   = ws;                       // 256*512   = 131072
  bf16* rWihb = ws + 131072;              // 768*256   = 196608
  bf16* rWhhb = ws + 327680;
  bf16* cWihb = ws + 524288;
  bf16* cWhhb = ws + 720896;
  bf16* W2b   = ws + 917504;              // 16*256    = 4096 ; total 921600 bf16

  const int nW1 = 131072, nG = 196608, nW2 = 4096;
  cvt_f32_to_bf16<<<(nW1 + 255) / 256, 256, 0, stream>>>(W1, W1b, nW1);
  cvt_f32_to_bf16<<<(nG + 255) / 256, 256, 0, stream>>>(rWih, rWihb, nG);
  cvt_f32_to_bf16<<<(nG + 255) / 256, 256, 0, stream>>>(rWhh, rWhhb, nG);
  cvt_f32_to_bf16<<<(nG + 255) / 256, 256, 0, stream>>>(cWih, cWihb, nG);
  cvt_f32_to_bf16<<<(nG + 255) / 256, 256, 0, stream>>>(cWhh, cWhhb, nG);
  cvt_f32_to_bf16<<<(nW2 + 255) / 256, 256, 0, stream>>>(W2, W2b, nW2);

  comm_agent_fused<<<NROWS / N_AGENTS, 256, 0, stream>>>(
      inputs, hidden, b1, rbih, rbhh, cbih, cbhh, b2,
      W1b, rWihb, rWhhb, cWihb, cWhhb, W2b, out);
}
